// BiLSTM_Biaffine_90881507983487
// MI455X (gfx1250) — matrix operand order repaired and checked by reading, not compiled, not run on hardware
//
#include <hip/hip_runtime.h>
#include <hip/hip_bf16.h>
#include <hip/hip_fp16.h>

// ---------------- problem constants ----------------
#define B_    32
#define L_    128
#define EMB_  300
#define EMBP  320     // EMB padded to mult of 32
#define H_    400     // hidden per direction
#define HP    416     // H padded to mult of 32
#define G4    1600    // 4*H (gates)
#define HID_  800     // BiLSTM output
#define FF    512     // FFNN
#define FFB   513     // FFNN + bias column
#define FFP   544     // FFB padded to mult of 32
#define NL    9
#define NROWS 4096    // B_*L_
#define NSQ   16384   // L_*L_

// tile counts
#define KT_XW   (EMBP / 32)   // 10
#define NT_G    (G4 / 16)     // 100
#define KT_HH   (HP / 32)     // 13
#define KT_FFN  (HID_ / 32)   // 25
#define NT_FFN  (FF / 16)     // 32
#define KT_BIA  (FFP / 32)    // 17
#define NT_BIA  (FFP / 16)    // 34

typedef __attribute__((ext_vector_type(16))) _Float16 v16h;
typedef __attribute__((ext_vector_type(8)))  _Float16 v8h;
typedef __attribute__((ext_vector_type(8)))  float    v8f;

// ---------------- WMMA fragment helpers (A/B k = 16*(lane/16) + i; C/D per ISA) ----------------
__device__ inline v8f zero8() {
  v8f z;
#pragma unroll
  for (int i = 0; i < 8; ++i) z[i] = 0.0f;
  return z;
}

__device__ inline v8f wmma16(v16h a, v16h b, v8f c) {
  return __builtin_amdgcn_wmma_f32_16x16x32_f16(false, a, false, b, (short)0, c,
                                                false, false);
}

// A tile 16x32 (MxK), row-major memory, row stride ld (elements, 16B-aligned rows).
// lane<16: row=lane, K 0..15 ; lane>=16: row=lane-16, K 16..31
__device__ inline v16h frag_a(const _Float16* p, int ld, int lane) {
  const int r  = lane & 15;
  const int kb = (lane & 16) ? 16 : 0;
  const _Float16* q = p + (size_t)r * ld + kb;
  v8h lo = *(const v8h*)(q);
  v8h hi = *(const v8h*)(q + 8);
  return __builtin_shufflevector(lo, hi, 0, 1, 2, 3, 4, 5, 6, 7,
                                 8, 9, 10, 11, 12, 13, 14, 15);
}

// B tile from fragment-packed memory: tile base -> lane-contiguous 16 halves
__device__ inline v16h frag_b_pk(const _Float16* tile, int lane) {
  return *(const v16h*)(tile + (size_t)lane * 16);
}

// B tile 32x16 where memory holds B^T row-major ([N][K], row stride ld, 32B-aligned)
__device__ inline v16h frag_bT(const _Float16* p, int ld, int lane) {
  const int c  = lane & 15;
  const int kb = (lane & 16) ? 16 : 0;
  return *(const v16h*)(p + (size_t)c * ld + kb);
}

// C/D element mapping: vgpr e -> row = e + (lane>=16 ? 8 : 0), col = lane&15
__device__ inline int cd_row(int lane, int e) { return e + ((lane & 16) ? 8 : 0); }
__device__ inline int cd_col(int lane)        { return lane & 15; }

// ---------------- prep kernels ----------------
// Embedding gather -> X f16 [L*B][EMBP], row = l*32 + b
__global__ void k_embed(const int* __restrict__ widx, const float* __restrict__ emb,
                        _Float16* __restrict__ X) {
  size_t t = (size_t)blockIdx.x * blockDim.x + threadIdx.x;
  if (t >= (size_t)NROWS * EMBP) return;
  int row = (int)(t / EMBP), e = (int)(t % EMBP);
  int l = row >> 5, b = row & 31;
  float v = 0.0f;
  if (e < EMB_) {
    int w = widx[b * L_ + l];
    v = emb[(size_t)w * EMB_ + e];
  }
  X[t] = (_Float16)v;
}

// Pack logical B[k][n] = src[n][k] (src is [sr][sc] = [Ntotal][Ktotal] f32, e.g. torch
// weight [out][in]) into WMMA-B fragment order: dst[((nt*KT + kt)*32 + lane)*16 + i].
__global__ void k_pack_bT(const float* __restrict__ src, _Float16* __restrict__ dst,
                          int sr, int sc, int KT, int NT) {
  size_t t = (size_t)blockIdx.x * blockDim.x + threadIdx.x;
  if (t >= (size_t)NT * KT * 512) return;
  int i    = (int)(t & 15);
  int lane = (int)((t >> 4) & 31);
  int kt   = (int)((t >> 9) % KT);
  int nt   = (int)((t >> 9) / KT);
  int k = kt * 32 + ((lane & 16) ? 16 : 0) + i;
  int n = nt * 16 + (lane & 15);
  float v = (k < sc && n < sr) ? src[(size_t)n * sc + k] : 0.0f;
  dst[t] = (_Float16)v;
}

// Pack logical B[k][n] = src[o][k][n] (W_biaffine [NL][513][513]) into fragment order:
// dst[(((o*NT + nt)*KT + kt)*32 + lane)*16 + i]
__global__ void k_pack_wb(const float* __restrict__ src, _Float16* __restrict__ dst) {
  size_t t = (size_t)blockIdx.x * blockDim.x + threadIdx.x;
  if (t >= (size_t)NL * NT_BIA * KT_BIA * 512) return;
  int i    = (int)(t & 15);
  int lane = (int)((t >> 4) & 31);
  size_t tile = t >> 9;
  int kt = (int)(tile % KT_BIA);
  int nt = (int)((tile / KT_BIA) % NT_BIA);
  int o  = (int)(tile / ((size_t)KT_BIA * NT_BIA));
  int k = kt * 32 + ((lane & 16) ? 16 : 0) + i;
  int n = nt * 16 + (lane & 15);
  float v = (k < FFB && n < FFB) ? src[((size_t)o * FFB + k) * FFB + n] : 0.0f;
  dst[t] = (_Float16)v;
}

// bsum[dir][j] = bih + bhh
__global__ void k_bias(const float* bih_f, const float* bhh_f,
                       const float* bih_b, const float* bhh_b,
                       float* __restrict__ bsum) {
  int t = blockIdx.x * blockDim.x + threadIdx.x;
  if (t >= 2 * G4) return;
  int d = t / G4, j = t % G4;
  bsum[t] = d ? (bih_b[j] + bhh_b[j]) : (bih_f[j] + bhh_f[j]);
}

// ---------------- GEMM: XW = x . Wih^T, stored C-fragment-packed ----------------
// XW_pk[((dir*256 + mt)*100 + nt)*32 + lane][8]
// grid (256, 25, 2), block 128 (4 waves, each one N tile)
__global__ void k_gemm_xw(const _Float16* __restrict__ X,
                          const _Float16* __restrict__ Wpk,  // [2][100][10][512]
                          float* __restrict__ XWpk) {
  const int lane = threadIdx.x & 31, wave = threadIdx.x >> 5;
  const int mt = blockIdx.x;
  const int nt = blockIdx.y * 4 + wave;
  const int dir = blockIdx.z;
  const _Float16* Wt = Wpk + (((size_t)dir * NT_G + nt) * KT_XW) * 512;
  const _Float16* A0 = X + (size_t)(mt * 16) * EMBP;
  v8f acc = zero8();
  for (int kk = 0; kk < KT_XW; ++kk) {
    v16h a = frag_a(A0 + kk * 32, EMBP, lane);
    v16h b = frag_b_pk(Wt + (size_t)kk * 512, lane);
    acc = wmma16(a, b, acc);
  }
  float* dst = XWpk + ((((size_t)dir * 256 + mt) * NT_G + nt) * 32 + lane) * 8;
  *(v8f*)dst = acc;
}

// ---------------- LSTM recurrence (1 workgroup per direction) ----------------
// block 512 threads = 16 waves; 50 tile-tasks (2 M-tiles x 25 hh-tiles),
// each task computes matched i/f/g/o gate tiles so the cell state lives in regs.
__global__ void k_lstm(const _Float16* __restrict__ Wpk,   // [2][100][13][512]
                       const float* __restrict__ bsum,     // [2][G4]
                       const float* __restrict__ XWpk,     // [2][256][100][32][8]
                       const int* __restrict__ widx,       // [B][L]
                       _Float16* __restrict__ enc) {       // [b*L+l][HID]
  __shared__ __align__(32) _Float16 hlds[32 * HP];
  const int dir = blockIdx.x;
  const int tid = threadIdx.x, lane = tid & 31, wave = tid >> 5;
  for (int i = tid; i < 32 * HP; i += blockDim.x) hlds[i] = (_Float16)0.0f;
  float creg[4][8], h2v[4][8], ov[4][8];
#pragma unroll
  for (int t4 = 0; t4 < 4; ++t4)
#pragma unroll
    for (int e = 0; e < 8; ++e) creg[t4][e] = 0.0f;
  const _Float16* Wd = Wpk + (size_t)dir * NT_G * KT_HH * 512;
  const float* bd = bsum + dir * G4;
  const float* XWd = XWpk + (size_t)dir * 256 * NT_G * 256;
  __syncthreads();

  for (int t = 0; t < L_; ++t) {
    const int l = dir ? (L_ - 1 - t) : t;
    const int ln = dir ? (L_ - 2 - t) : (t + 1);  // next timestep (prefetch)
#pragma unroll
    for (int t4 = 0; t4 < 4; ++t4) {
      const int task = wave + 16 * t4;
      if (task < 50) {
        const int mt = task / 25, nt = task % 25;
        v8f acc[4];
#pragma unroll
        for (int g = 0; g < 4; ++g) {
          const int gnt = g * 25 + nt;  // global n-tile 0..99
          const float* xb =
              XWd + (((size_t)(l * 2 + mt) * NT_G + gnt) * 32 + lane) * 8;
          float bias = bd[g * H_ + nt * 16 + (lane & 15)];
          acc[g] = *(const v8f*)xb + bias;
          if (ln >= 0 && ln < L_) {
            __builtin_prefetch(
                XWd + (((size_t)(ln * 2 + mt) * NT_G + gnt) * 32 + lane) * 8, 0, 1);
          }
        }
        for (int kk = 0; kk < KT_HH; ++kk) {
          v16h a = frag_a(hlds + (size_t)(mt * 16) * HP + kk * 32, HP, lane);
#pragma unroll
          for (int g = 0; g < 4; ++g) {
            const int gnt = g * 25 + nt;
            v16h bb = frag_b_pk(Wd + ((size_t)gnt * KT_HH + kk) * 512, lane);
            acc[g] = wmma16(a, bb, acc[g]);
          }
        }
#pragma unroll
        for (int e = 0; e < 8; ++e) {
          int brow = mt * 16 + cd_row(lane, e);
          int hh   = nt * 16 + cd_col(lane);
          float gi = 1.0f / (1.0f + __expf(-acc[0][e]));
          float gf = 1.0f / (1.0f + __expf(-acc[1][e]));
          float gg = tanhf(acc[2][e]);
          float go = 1.0f / (1.0f + __expf(-acc[3][e]));
          float cn = gf * creg[t4][e] + gi * gg;
          float hn = go * tanhf(cn);
          bool  m  = widx[brow * L_ + l] > 0;
          float hp = (float)hlds[brow * HP + hh];
          creg[t4][e] = m ? cn : creg[t4][e];
          h2v[t4][e]  = m ? hn : hp;
          ov[t4][e]   = m ? hn : 0.0f;
        }
      }
    }
    __syncthreads();  // everyone finished reading h_prev
#pragma unroll
    for (int t4 = 0; t4 < 4; ++t4) {
      const int task = wave + 16 * t4;
      if (task < 50) {
        const int mt = task / 25, nt = task % 25;
#pragma unroll
        for (int e = 0; e < 8; ++e) {
          int brow = mt * 16 + cd_row(lane, e);
          int hh   = nt * 16 + cd_col(lane);
          hlds[brow * HP + hh] = (_Float16)h2v[t4][e];
          enc[((size_t)brow * L_ + l) * HID_ + dir * H_ + hh] = (_Float16)ov[t4][e];
        }
      }
    }
    __syncthreads();  // h writes visible before next step
  }
}

// ---------------- FFNN heads: out = f16(enc @ W^T + b) into [NROWS][FFP] ----------------
// grid (256, 8), block 128
__global__ void k_gemm_ffnn(const _Float16* __restrict__ enc,
                            const _Float16* __restrict__ Wpk,  // [32][25][512]
                            const float* __restrict__ bias,
                            _Float16* __restrict__ out) {
  const int lane = threadIdx.x & 31, wave = threadIdx.x >> 5;
  const int mt = blockIdx.x;
  const int nt = blockIdx.y * 4 + wave;
  const _Float16* Wt = Wpk + (size_t)nt * KT_FFN * 512;
  const _Float16* A0 = enc + (size_t)(mt * 16) * HID_;
  v8f acc = zero8();
  for (int kk = 0; kk < KT_FFN; ++kk) {
    v16h a = frag_a(A0 + kk * 32, HID_, lane);
    v16h b = frag_b_pk(Wt + (size_t)kk * 512, lane);
    acc = wmma16(a, b, acc);
  }
  float bv = bias[nt * 16 + (lane & 15)];
#pragma unroll
  for (int e = 0; e < 8; ++e) {
    int row = mt * 16 + cd_row(lane, e);
    int col = nt * 16 + cd_col(lane);
    out[(size_t)row * FFP + col] = (_Float16)(acc[e] + bv);
  }
}

// append bias-1 column (512) and zero pad (513..543) for x1 and y1
__global__ void k_tail(_Float16* __restrict__ x1, _Float16* __restrict__ y1) {
  int t = blockIdx.x * blockDim.x + threadIdx.x;
  if (t >= NROWS * 32) return;
  int row = t >> 5, c = 512 + (t & 31);
  _Float16 v = (_Float16)((c == 512) ? 1.0f : 0.0f);
  x1[(size_t)row * FFP + c] = v;
  y1[(size_t)row * FFP + c] = v;
}

// ---------------- biaffine stage 1: tmp[b,o,x,:] = x1[b,x,:] @ W[o] ----------------
// grid (8, 34, 288=b*9), block 32
__global__ void k_gemm_bia1(const _Float16* __restrict__ x1,
                            const _Float16* __restrict__ Wbpk,  // [9][34][17][512]
                            _Float16* __restrict__ tmp) {       // [288*128][FFP]
  const int lane = threadIdx.x;
  const int mt = blockIdx.x, nt = blockIdx.y, bo = blockIdx.z;
  const int b = bo / NL, o = bo % NL;
  const _Float16* A0 = x1 + (size_t)(b * L_ + mt * 16) * FFP;
  const _Float16* Wt = Wbpk + (((size_t)o * NT_BIA + nt) * KT_BIA) * 512;
  v8f acc = zero8();
  for (int kk = 0; kk < KT_BIA; ++kk) {
    v16h a  = frag_a(A0 + kk * 32, FFP, lane);
    v16h bb = frag_b_pk(Wt + (size_t)kk * 512, lane);
    acc = wmma16(a, bb, acc);
  }
#pragma unroll
  for (int e = 0; e < 8; ++e) {
    int row = mt * 16 + cd_row(lane, e);
    int col = nt * 16 + cd_col(lane);
    tmp[((size_t)bo * L_ + row) * FFP + col] = (_Float16)acc[e];
  }
}

// ---------------- biaffine stage 2: score[b,x,y,o] = tmp[b,o,x,:] . y1[b,y,:] ----------------
// grid (8, 8, 288), block 32
__global__ void k_gemm_bia2(const _Float16* __restrict__ tmp,
                            const _Float16* __restrict__ y1,
                            float* __restrict__ score) {  // [B][L][L][NL]
  const int lane = threadIdx.x;
  const int xt = blockIdx.x, yt = blockIdx.y, bo = blockIdx.z;
  const int b = bo / NL, o = bo % NL;
  const _Float16* A0 = tmp + ((size_t)bo * L_ + xt * 16) * FFP;
  const _Float16* Bt = y1 + (size_t)(b * L_ + yt * 16) * FFP;  // rows = y, cols = k
  v8f acc = zero8();
  for (int kk = 0; kk < KT_BIA; ++kk) {
    v16h a  = frag_a(A0 + kk * 32, FFP, lane);
    v16h bb = frag_bT(Bt + kk * 32, FFP, lane);
    acc = wmma16(a, bb, acc);
  }
#pragma unroll
  for (int e = 0; e < 8; ++e) {
    int x = xt * 16 + cd_row(lane, e);
    int y = yt * 16 + cd_col(lane);
    score[(((size_t)b * L_ + x) * L_ + y) * NL + o] = acc[e];
  }
}

// ---------------- argmax / max over labels ----------------
__global__ void k_argmax(const float* __restrict__ score,
                         float* __restrict__ pscore, int* __restrict__ pans) {
  size_t t = (size_t)blockIdx.x * blockDim.x + threadIdx.x;
  if (t >= (size_t)B_ * NSQ) return;
  const float* s = score + t * NL;
  float best = s[0];
  int bi = 0;
#pragma unroll
  for (int o = 1; o < NL; ++o) {
    float v = s[o];
    if (v > best) { best = v; bi = o; }
  }
  pscore[t] = best;
  pans[t] = bi;
}

// ---------------- per-sentence greedy decode ----------------
__device__ inline void rangeMask128(int lo, int hi, unsigned long long& m0,
                                    unsigned long long& m1) {
  // set bits lo..hi inclusive, 0 <= lo <= hi < 128
  m0 = 0ull; m1 = 0ull;
  if (lo < 64) {
    int h = (hi < 64) ? hi : 63;
    unsigned long long m = (h >= 63) ? ~0ull : ((1ull << (h + 1)) - 1ull);
    m0 = m & (~0ull << lo);
  }
  if (hi >= 64) {
    int l = (lo >= 64) ? (lo - 64) : 0;
    int h = hi - 64;
    unsigned long long m = (h >= 63) ? ~0ull : ((1ull << (h + 1)) - 1ull);
    m1 = m & (~0ull << l);
  }
}

__global__ void k_decode(const float* __restrict__ pscore,
                         const int* __restrict__ pans,
                         const int* __restrict__ labels,
                         float* __restrict__ out) {
  extern __shared__ unsigned long long key[];  // NSQ u64 keys (128 KB)
  const int b = blockIdx.x, tid = threadIdx.x, NT = blockDim.x;
  const float* ps = pscore + (size_t)b * NSQ;
  const int* pa = pans + (size_t)b * NSQ;
  const int* lb = labels + (size_t)b * NSQ;
  float* ob = out + (size_t)b * NSQ;

  for (int i = tid; i < NSQ; i += NT) {
    ob[i] = 1.0f;  // NON_ENTITY
    float sc = ps[i];
    bool valid = (pa[i] != 1) && (lb[i] > 0);
    if (!valid) sc = -__builtin_huge_valf();
    unsigned u = __float_as_uint(sc);
    u = (u & 0x80000000u) ? ~u : (u | 0x80000000u);          // monotone ascending map
    key[i] = ((unsigned long long)(~u) << 32) | (unsigned)i; // score desc, idx asc
  }
  __syncthreads();

  // bitonic sort ascending on u64 keys
  for (int kk = 2; kk <= NSQ; kk <<= 1) {
    for (int jj = kk >> 1; jj > 0; jj >>= 1) {
      for (int i = tid; i < NSQ; i += NT) {
        int ixj = i ^ jj;
        if (ixj > i) {
          bool up = (i & kk) == 0;
          unsigned long long a = key[i], c = key[ixj];
          if ((a > c) == up) { key[i] = c; key[ixj] = a; }
        }
      }
      __syncthreads();
    }
  }

  if (tid == 0) {
    unsigned long long s0 = 0, s1 = 0, in0 = 0, in1 = 0;
    for (int m = 0; m < NSQ; ++m) {
      unsigned idx = (unsigned)(key[m] & 0xFFFFFFFFu);
      int i = idx >> 7, j = idx & 127;
      int a = pa[idx];
      bool v = (a != 1) && (lb[idx] > 0);
      bool conflict = false;
      if (i <= j) {
        unsigned long long m0, m1;
        rangeMask128(i, j, m0, m1);
        conflict = ((s0 & m0) | (s1 & m1)) != 0ull;
      }
      conflict = conflict || (((i < 64 ? (in0 >> i) : (in1 >> (i - 64))) & 1ull) != 0ull);
      if (v && !conflict) {
        if (i < 64) s0 |= (1ull << i); else s1 |= (1ull << (i - 64));
        if (i <= j) {
          unsigned long long m0, m1;
          rangeMask128(i, j, m0, m1);
          in0 |= m0; in1 |= m1;
        }
        ob[idx] = (float)a;
      }
    }
  }
}

// ---------------- host launch ----------------
extern "C" void kernel_launch(void* const* d_in, const int* in_sizes, int n_in,
                              void* d_out, int out_size, void* d_ws, size_t ws_size,
                              hipStream_t stream) {
  (void)in_sizes; (void)n_in; (void)out_size; (void)ws_size;
  const int*   widx   = (const int*)d_in[0];
  const int*   labels = (const int*)d_in[1];
  const float* emb    = (const float*)d_in[2];
  const float* Wih_f  = (const float*)d_in[3];
  const float* Whh_f  = (const float*)d_in[4];
  const float* bih_f  = (const float*)d_in[5];
  const float* bhh_f  = (const float*)d_in[6];
  const float* Wih_b  = (const float*)d_in[7];
  const float* Whh_b  = (const float*)d_in[8];
  const float* bih_b  = (const float*)d_in[9];
  const float* bhh_b  = (const float*)d_in[10];
  const float* W_start = (const float*)d_in[11];
  const float* b_start = (const float*)d_in[12];
  const float* W_end   = (const float*)d_in[13];
  const float* b_end   = (const float*)d_in[14];
  const float* W_bia   = (const float*)d_in[15];
  float* out = (float*)d_out;

  // workspace carve-out (256B aligned blocks)
  char* p = (char*)d_ws;
  auto alloc = [&](size_t bytes) -> void* {
    void* r = (void*)p;
    p += (bytes + 255) & ~(size_t)255;
    return r;
  };
  _Float16* X      = (_Float16*)alloc((size_t)NROWS * EMBP * 2);
  _Float16* WihPk  = (_Float16*)alloc((size_t)2 * NT_G * KT_XW * 512 * 2);
  _Float16* WhhPk  = (_Float16*)alloc((size_t)2 * NT_G * KT_HH * 512 * 2);
  float*    bsum   = (float*)alloc((size_t)2 * G4 * 4);
  float*    XWpk   = (float*)alloc((size_t)2 * 256 * NT_G * 256 * 4);
  _Float16* enc    = (_Float16*)alloc((size_t)NROWS * HID_ * 2);
  _Float16* WsPk   = (_Float16*)alloc((size_t)NT_FFN * KT_FFN * 512 * 2);
  _Float16* WePk   = (_Float16*)alloc((size_t)NT_FFN * KT_FFN * 512 * 2);
  _Float16* x1     = (_Float16*)alloc((size_t)NROWS * FFP * 2);
  _Float16* y1     = (_Float16*)alloc((size_t)NROWS * FFP * 2);
  _Float16* WbPk   = (_Float16*)alloc((size_t)NL * NT_BIA * KT_BIA * 512 * 2);
  _Float16* tmp    = (_Float16*)alloc((size_t)B_ * NL * L_ * FFP * 2);
  float*    score  = (float*)alloc((size_t)B_ * NSQ * NL * 4);
  float*    psc    = (float*)alloc((size_t)B_ * NSQ * 4);
  int*      pan    = (int*)alloc((size_t)B_ * NSQ * 4);

  // ---- prep / conversion ----
  {
    size_t n = (size_t)NROWS * EMBP;
    k_embed<<<dim3((unsigned)((n + 255) / 256)), 256, 0, stream>>>(widx, emb, X);
  }
  {
    size_t n = (size_t)NT_G * KT_XW * 512;
    k_pack_bT<<<dim3((unsigned)((n + 255) / 256)), 256, 0, stream>>>(
        Wih_f, WihPk, G4, EMB_, KT_XW, NT_G);
    k_pack_bT<<<dim3((unsigned)((n + 255) / 256)), 256, 0, stream>>>(
        Wih_b, WihPk + n, G4, EMB_, KT_XW, NT_G);
  }
  {
    size_t n = (size_t)NT_G * KT_HH * 512;
    k_pack_bT<<<dim3((unsigned)((n + 255) / 256)), 256, 0, stream>>>(
        Whh_f, WhhPk, G4, H_, KT_HH, NT_G);
    k_pack_bT<<<dim3((unsigned)((n + 255) / 256)), 256, 0, stream>>>(
        Whh_b, WhhPk + n, G4, H_, KT_HH, NT_G);
  }
  {
    size_t n = (size_t)NT_FFN * KT_FFN * 512;
    k_pack_bT<<<dim3((unsigned)((n + 255) / 256)), 256, 0, stream>>>(
        W_start, WsPk, FF, HID_, KT_FFN, NT_FFN);
    k_pack_bT<<<dim3((unsigned)((n + 255) / 256)), 256, 0, stream>>>(
        W_end, WePk, FF, HID_, KT_FFN, NT_FFN);
  }
  {
    size_t n = (size_t)NL * NT_BIA * KT_BIA * 512;
    k_pack_wb<<<dim3((unsigned)((n + 255) / 256)), 256, 0, stream>>>(W_bia, WbPk);
  }
  k_bias<<<dim3((2 * G4 + 255) / 256), 256, 0, stream>>>(bih_f, bhh_f, bih_b, bhh_b,
                                                         bsum);

  // ---- compute ----
  k_gemm_xw<<<dim3(NROWS / 16, NT_G / 4, 2), 128, 0, stream>>>(X, WihPk, XWpk);
  k_lstm<<<dim3(2), 512, 0, stream>>>(WhhPk, bsum, XWpk, widx, enc);
  k_gemm_ffnn<<<dim3(NROWS / 16, NT_FFN / 4), 128, 0, stream>>>(enc, WsPk, b_start, x1);
  k_gemm_ffnn<<<dim3(NROWS / 16, NT_FFN / 4), 128, 0, stream>>>(enc, WePk, b_end, y1);
  k_tail<<<dim3((NROWS * 32 + 255) / 256), 256, 0, stream>>>(x1, y1);
  k_gemm_bia1<<<dim3(L_ / 16, NT_BIA, B_ * NL), 32, 0, stream>>>(x1, WbPk, tmp);
  k_gemm_bia2<<<dim3(L_ / 16, L_ / 16, B_ * NL), 32, 0, stream>>>(tmp, y1, score);
  {
    size_t n = (size_t)B_ * NSQ;
    k_argmax<<<dim3((unsigned)((n + 255) / 256)), 256, 0, stream>>>(score, psc, pan);
  }
  k_decode<<<dim3(B_), 1024, NSQ * sizeof(unsigned long long), stream>>>(psc, pan,
                                                                         labels, out);
}
